// NTXentLoss_67422396612651
// MI455X (gfx1250) — matrix operand order repaired and checked by reading, not compiled, not run on hardware
//
#include <hip/hip_runtime.h>
#include <hip/hip_fp16.h>
#include <math.h>

typedef __attribute__((ext_vector_type(16))) _Float16 v16h;
typedef __attribute__((ext_vector_type(8)))  _Float16 v8h;
typedef __attribute__((ext_vector_type(4)))  _Float16 v4h;
typedef __attribute__((ext_vector_type(8)))  float    v8f;

#define NROWS      8192          // 2 * B
#define BHALF      4096
#define DIM        512
#define LDS_STRIDE 520           // 512 + 8-half pad -> A-frag lane reads hit distinct banks

union AFrag { struct { v8h lo, hi; } p; v16h v; };

// ---------------------------------------------------------------------------
// Kernel 1: L2-normalize each row of concat(z_i, z_j); emit f16 row-major (zh)
// and f16 transposed (zt) copies; WMMA A and B fragments are both read from zh.
// ---------------------------------------------------------------------------
__global__ __launch_bounds__(128)
void nt_normalize(const float* __restrict__ zi, const float* __restrict__ zj,
                  _Float16* __restrict__ zh, _Float16* __restrict__ zt)
{
    const int row = blockIdx.x;
    const int tid = threadIdx.x;
    const float* src = (row < BHALF) ? (zi + (size_t)row * DIM)
                                     : (zj + (size_t)(row - BHALF) * DIM);
    float4 v = ((const float4*)src)[tid];                 // 128 thr * 4 = 512
    float ss = v.x * v.x + v.y * v.y + v.z * v.z + v.w * v.w;
    #pragma unroll
    for (int off = 16; off > 0; off >>= 1)
        ss += __shfl_xor(ss, off, 32);

    __shared__ float part[4];
    __shared__ float invs;
    if ((tid & 31) == 0) part[tid >> 5] = ss;
    __syncthreads();
    if (tid == 0) {
        float t = part[0] + part[1] + part[2] + part[3];
        invs = 1.0f / fmaxf(sqrtf(t), 1e-12f);            // torch F.normalize eps
    }
    __syncthreads();
    const float inv = invs;

    const int k = tid * 4;
    _Float16 h0 = (_Float16)(v.x * inv);
    _Float16 h1 = (_Float16)(v.y * inv);
    _Float16 h2 = (_Float16)(v.z * inv);
    _Float16 h3 = (_Float16)(v.w * inv);

    *(v4h*)(zh + (size_t)row * DIM + k) = (v4h){h0, h1, h2, h3};

    zt[(size_t)(k + 0) * NROWS + row] = h0;
    zt[(size_t)(k + 1) * NROWS + row] = h1;
    zt[(size_t)(k + 2) * NROWS + row] = h2;
    zt[(size_t)(k + 3) * NROWS + row] = h3;
}

// ---------------------------------------------------------------------------
// Kernel 2: one block per 32-row strip (two 16-row M-tiles in LDS). Each wave
// runs a 2x2 register tile per K-step: 2 A-frags (LDS) x 2 B-frags (L2) feed
// 4 v_wmma -> 1 ds_load_b128 + 1 global_load_b128 per WMMA.
// Fixed-shift softmax: sim/TEMP in [-2,2] (rows normalized) => shift 2 is a
// safe stabilizer; hot path is a branchless s += exp(v-2). Diagonal tile
// (c==Mt) and positive tile (c==Mt+-256) are scalar-uniform fixups.
// ---------------------------------------------------------------------------
__global__ __launch_bounds__(256)
void nt_sim_lse(const _Float16* __restrict__ zh, const _Float16* __restrict__ zt,
                float* __restrict__ partials)
{
    __shared__ _Float16 As[32 * LDS_STRIDE];              // 33,280 B
    __shared__ float sS[8][32], sP[8][32];

    const int r    = blockIdx.x;        // 0..255, rows 32r .. 32r+31
    const int tid  = threadIdx.x;
    const int wave = tid >> 5;
    const int lane = tid & 31;
    const int half = lane >> 4;         // 0: lanes 0-15, 1: lanes 16-31
    const int l16  = lane & 15;

    // Stage 32x512 f16 A strip into padded LDS.
    {
        const _Float16* src = zh + (size_t)r * 32 * DIM;
        for (int i = tid; i < 2048; i += 256) {           // 2048 chunks of 8 halves
            int row = i >> 6;
            int kc  = (i & 63) << 3;
            *(uint4*)(&As[row * LDS_STRIDE + kc]) = *(const uint4*)(src + row * DIM + kc);
        }
    }
    __syncthreads();

    float rs[2][8], pp[2][8];
    #pragma unroll
    for (int mt = 0; mt < 2; ++mt)
        #pragma unroll
        for (int i = 0; i < 8; ++i) { rs[mt][i] = 0.0f; pp[mt][i] = 0.0f; }

    // A-fragments: lane reads LDS row l16 (tile 0) and l16+16 (tile 1),
    // K offsets {half*8..+7} and {+16..+23} per the 16-bit A layout.
    const _Float16* aRow0 = &As[l16 * LDS_STRIDE + half * 8];
    const _Float16* aRow1 = aRow0 + 16 * LDS_STRIDE;
    // B-fragment: lane holds column l16 (zh row), K offsets {half*8..+7} and {+16..+23}.
    const int lk = l16 + 16 * half;

    const int mt0   = 2 * r;                              // global M-tile ids
    const int mt1   = 2 * r + 1;
    const int dpos  = (r < 128) ? 256 : -256;             // whole strip same half
    const int posT0 = mt0 + dpos;
    const int posT1 = mt1 + dpos;

    for (int t = 0; t < 32; ++t) {
        const int c0 = t * 16 + wave * 2;                 // this wave's tile pair
        const _Float16* bBase = zh + (size_t)(c0 * 16 + l16) * DIM + half * 8;
        v8f acc00 = {}, acc01 = {}, acc10 = {}, acc11 = {};
        #pragma unroll 2
        for (int s = 0; s < 16; ++s) {                    // K = 512 in steps of 32
            const int kb = s * 32;
            AFrag a0, a1;
            a0.p.lo = *(const v8h*)(aRow0 + kb);
            a0.p.hi = *(const v8h*)(aRow0 + kb + 16);
            a1.p.lo = *(const v8h*)(aRow1 + kb);
            a1.p.hi = *(const v8h*)(aRow1 + kb + 16);
            const _Float16* bk = bBase + kb;
            AFrag b0f, b1f;
            b0f.p.lo = *(const v8h*)(bk);
            b0f.p.hi = *(const v8h*)(bk + 16);
            b1f.p.lo = *(const v8h*)(bk + 16 * DIM);
            b1f.p.hi = *(const v8h*)(bk + 16 * DIM + 16);
            v16h b0 = b0f.v;
            v16h b1 = b1f.v;
            acc00 = __builtin_amdgcn_wmma_f32_16x16x32_f16(
                        false, a0.v, false, b0, (short)0, acc00, false, false);
            acc01 = __builtin_amdgcn_wmma_f32_16x16x32_f16(
                        false, a0.v, false, b1, (short)0, acc01, false, false);
            acc10 = __builtin_amdgcn_wmma_f32_16x16x32_f16(
                        false, a1.v, false, b0, (short)0, acc10, false, false);
            acc11 = __builtin_amdgcn_wmma_f32_16x16x32_f16(
                        false, a1.v, false, b1, (short)0, acc11, false, false);
        }

        #pragma unroll
        for (int mt = 0; mt < 2; ++mt) {
            const int Mt = (mt == 0) ? mt0 : mt1;
            const int pT = (mt == 0) ? posT0 : posT1;
            #pragma unroll
            for (int j = 0; j < 2; ++j) {
                const v8f acc = (mt == 0) ? (j == 0 ? acc00 : acc01)
                                          : (j == 0 ? acc10 : acc11);
                const int c = c0 + j;

                // Hot path: branchless shifted-exp accumulation.
                #pragma unroll
                for (int i = 0; i < 8; ++i)
                    rs[mt][i] += __expf(fmaf(acc[i], 2.0f, -2.0f));

                // Rare scalar-uniform fixups.
                if (c == Mt) {                            // cancel diagonal term
                    #pragma unroll
                    for (int i = 0; i < 8; ++i)
                        rs[mt][i] -= (l16 == i + 8 * half)
                                         ? __expf(fmaf(acc[i], 2.0f, -2.0f)) : 0.0f;
                }
                if (c == pT) {                            // capture positive sim
                    #pragma unroll
                    for (int i = 0; i < 8; ++i)
                        pp[mt][i] += (l16 == i + 8 * half) ? (acc[i] * 2.0f) : 0.0f;
                }
            }
        }
    }

    // Sum (s, p) across the 16 lanes sharing each row.
    #pragma unroll
    for (int mt = 0; mt < 2; ++mt) {
        #pragma unroll
        for (int i = 0; i < 8; ++i) {
            float s = rs[mt][i], p = pp[mt][i];
            #pragma unroll
            for (int off = 1; off < 16; off <<= 1) {
                s += __shfl_xor(s, off, 32);
                p += __shfl_xor(p, off, 32);
            }
            if (l16 == 0) {
                const int row32 = mt * 16 + i + 8 * half;
                sS[wave][row32] = s; sP[wave][row32] = p;
            }
        }
    }
    __syncthreads();

    // Wave 0: 32 lanes <-> 32 rows; merge wave partials, emit strip partial.
    if (wave == 0) {
        float s = 0.0f, p = 0.0f;
        #pragma unroll
        for (int w = 0; w < 8; ++w) { s += sS[w][lane]; p += sP[w][lane]; }
        float contrib = (2.0f + __logf(s)) - p;           // lse_row - pos_row
        #pragma unroll
        for (int off = 16; off > 0; off >>= 1)
            contrib += __shfl_xor(contrib, off, 32);
        if (lane == 0) partials[r] = contrib;
    }
}

// ---------------------------------------------------------------------------
// Kernel 3: deterministic fixed-order tree reduction of the 256 strip partials.
// ---------------------------------------------------------------------------
__global__ __launch_bounds__(128)
void nt_final(const float* __restrict__ partials, float* __restrict__ out)
{
    __shared__ float red[128];
    const int tid = threadIdx.x;
    red[tid] = partials[tid] + partials[tid + 128];
    __syncthreads();
    for (int s = 64; s > 0; s >>= 1) {
        if (tid < s) red[tid] += red[tid + s];
        __syncthreads();
    }
    if (tid == 0) out[0] = red[0] * (1.0f / (float)NROWS);
}

extern "C" void kernel_launch(void* const* d_in, const int* in_sizes, int n_in,
                              void* d_out, int out_size, void* d_ws, size_t ws_size,
                              hipStream_t stream)
{
    const float* zi = (const float*)d_in[0];
    const float* zj = (const float*)d_in[1];

    _Float16* zh       = (_Float16*)d_ws;                       // 8 MB row-major f16
    _Float16* zt       = zh + (size_t)NROWS * DIM;              // 8 MB transposed f16
    float*    partials = (float*)(zt + (size_t)NROWS * DIM);    // 256 floats

    nt_normalize<<<NROWS, 128, 0, stream>>>(zi, zj, zh, zt);
    nt_sim_lse <<<256,   256, 0, stream>>>(zh, zt, partials);
    nt_final   <<<1,     128, 0, stream>>>(partials, (float*)d_out);
}
